// AttentiveProtoFusion_23613730193836
// MI455X (gfx1250) — matrix operand order repaired and checked by reading, not compiled, not run on hardware
//
#include <hip/hip_runtime.h>
#include <hip/hip_bf16.h>

// Problem constants (from reference)
#define Bn 4
#define Sn 512
#define Pn 32
#define DSENT 1024
#define DCTX 768
#define BS (Bn * Sn)            // 2048 rows

typedef __attribute__((ext_vector_type(16))) __bf16 v16bf;
typedef __attribute__((ext_vector_type(8)))  float  v8f;

static __device__ __forceinline__ __bf16 f2bf(float f) { return (__bf16)f; }

#if __has_builtin(__builtin_amdgcn_sched_barrier)
#define SCHED_FENCE() __builtin_amdgcn_sched_barrier(0)
#else
#define SCHED_FENCE()
#endif

// ---------------------------------------------------------------------------
// Prep kernels: fp32 -> bf16 conversions (and transpose for Wq)
// ---------------------------------------------------------------------------
__global__ __launch_bounds__(256) void cvt_bf16_kernel(const float* __restrict__ in,
                                                       __bf16* __restrict__ out, int n) {
    int i = blockIdx.x * 256 + threadIdx.x;
    if (i < n) out[i] = f2bf(in[i]);
}

// out[n*rows + k] = in[k*cols + n]   (store W as [n][k], k contiguous)
__global__ __launch_bounds__(256) void transpose_cvt_kernel(const float* __restrict__ in,
                                                            __bf16* __restrict__ out,
                                                            int rows, int cols) {
    int i = blockIdx.x * 256 + threadIdx.x;
    if (i < rows * cols) {
        int n = i / rows;
        int k = i % rows;
        out[i] = f2bf(in[k * cols + n]);
    }
}

// ---------------------------------------------------------------------------
// WMMA fragment loaders (16x16x32 bf16, wave32 layouts per CDNA5 ISA 7.12.2)
// A (16x32, row-major, lda elems): lane l: M=l&15, kh=l>>4
//   VGPR0-3 : K = kh*8 + [0..7]     VGPR4-7 : K = 16 + kh*8 + [0..7]
// B (32x16) from Bt[n][k] (k contiguous, ldb elems): lane l: N=l&15, kh=l>>4
//   VGPR0-3 : K = kh*8 + [0..7]     VGPR4-7 : K = 16 + kh*8 + [0..7]
// ---------------------------------------------------------------------------
static __device__ __forceinline__ v16bf load_frag_a(const __bf16* __restrict__ A, int lda) {
    int lane = threadIdx.x & 31;
    int m = lane & 15;
    int kh = lane >> 4;
    const __bf16* p0 = A + m * lda + kh * 8;
    union { v16bf v; uint4 q[2]; } u;
    u.q[0] = *(const uint4*)(p0);
    u.q[1] = *(const uint4*)(p0 + 16);
    return u.v;
}

static __device__ __forceinline__ v16bf load_frag_b(const __bf16* __restrict__ Bt, int ldb) {
    int lane = threadIdx.x & 31;
    int n = lane & 15;
    int kh = lane >> 4;
    const __bf16* p0 = Bt + n * ldb + kh * 8;
    union { v16bf v; uint4 q[2]; } u;
    u.q[0] = *(const uint4*)(p0);
    u.q[1] = *(const uint4*)(p0 + 16);
    return u.v;
}

// One 16(M) x 64(N) strip per wave, software-pipelined with sched fences so
// the compiler cannot sink the next k-step's 10 loads back under the WMMAs.
// Final order per step:   [loads k+1] | fence | [4x WMMA on k] | fence
// => every WMMA waits only on year-old loads (partial s_wait_loadcnt), while
// 10 b128 loads stream in flight behind the matrix pipe.
template <int K>
static __device__ __forceinline__ void gemm_strip(const __bf16* __restrict__ A,
                                                  const __bf16* __restrict__ Bt,
                                                  int m0, int n0, v8f c[4]) {
    const __bf16* Ab = A + m0 * K;
    const __bf16* Bb = Bt + n0 * K;

    v16bf a = load_frag_a(Ab, K);
    v16bf b0 = load_frag_b(Bb + 0 * 16 * K, K);
    v16bf b1 = load_frag_b(Bb + 1 * 16 * K, K);
    v16bf b2 = load_frag_b(Bb + 2 * 16 * K, K);
    v16bf b3 = load_frag_b(Bb + 3 * 16 * K, K);

    #pragma unroll
    for (int k0 = 0; k0 < K; k0 += 32) {
        v16bf an, bn0, bn1, bn2, bn3;
        if (k0 + 32 < K) {                     // compile-time after full unroll
            an  = load_frag_a(Ab + k0 + 32, K);
            bn0 = load_frag_b(Bb + 0 * 16 * K + k0 + 32, K);
            bn1 = load_frag_b(Bb + 1 * 16 * K + k0 + 32, K);
            bn2 = load_frag_b(Bb + 2 * 16 * K + k0 + 32, K);
            bn3 = load_frag_b(Bb + 3 * 16 * K + k0 + 32, K);
        }
        SCHED_FENCE();
        c[0] = __builtin_amdgcn_wmma_f32_16x16x32_bf16(false, a, false, b0,
                                                       (short)0, c[0], false, false);
        c[1] = __builtin_amdgcn_wmma_f32_16x16x32_bf16(false, a, false, b1,
                                                       (short)0, c[1], false, false);
        c[2] = __builtin_amdgcn_wmma_f32_16x16x32_bf16(false, a, false, b2,
                                                       (short)0, c[2], false, false);
        c[3] = __builtin_amdgcn_wmma_f32_16x16x32_bf16(false, a, false, b3,
                                                       (short)0, c[3], false, false);
        SCHED_FENCE();
        if (k0 + 32 < K) {
            a = an; b0 = bn0; b1 = bn1; b2 = bn2; b3 = bn3;
        }
    }
}

// ---------------------------------------------------------------------------
// GEMM Q: Qbf[m][n] = bf16( sum_k sent_bf[m][k] * WqT[n][k] + bq[n] )
// M=2048, K=1024, N=768.  1536 strips (128 x 12), 8 waves per block.
// ---------------------------------------------------------------------------
__global__ __launch_bounds__(256) void gemm_q_kernel(const __bf16* __restrict__ A,
                                                     const __bf16* __restrict__ Bt,
                                                     const float* __restrict__ bias,
                                                     __bf16* __restrict__ C) {
    int wave = threadIdx.x >> 5;
    int strip = blockIdx.x * 8 + wave;
    int m0 = (strip / 12) << 4;
    int n0 = (strip % 12) << 6;
    int lane = threadIdx.x & 31;

    v8f c[4] = {{}, {}, {}, {}};
    gemm_strip<DSENT>(A, Bt, m0, n0, c);

    int n = lane & 15;
    int kh = lane >> 4;
    #pragma unroll
    for (int j = 0; j < 4; ++j) {
        float bb = bias[n0 + j * 16 + n];
        #pragma unroll
        for (int r = 0; r < 8; ++r) {
            int row = m0 + kh * 8 + r;          // lanes16-31 -> M = 8 + r
            C[row * DCTX + n0 + j * 16 + n] = f2bf(c[j][r] + bb);
        }
    }
}

// ---------------------------------------------------------------------------
// GEMM R: R[m][c] = sum_d Qbf[m][d] * Wk_bf[c][d]   (= Q @ Wk^T, fp32 out)
// M=2048, K=768, N=768.  1536 strips (128 x 12).
// ---------------------------------------------------------------------------
__global__ __launch_bounds__(256) void gemm_r_kernel(const __bf16* __restrict__ A,
                                                     const __bf16* __restrict__ Bt,
                                                     float* __restrict__ C) {
    int wave = threadIdx.x >> 5;
    int strip = blockIdx.x * 8 + wave;
    int m0 = (strip / 12) << 4;
    int n0 = (strip % 12) << 6;
    int lane = threadIdx.x & 31;

    v8f c[4] = {{}, {}, {}, {}};
    gemm_strip<DCTX>(A, Bt, m0, n0, c);

    int n = lane & 15;
    int kh = lane >> 4;
    #pragma unroll
    for (int j = 0; j < 4; ++j) {
        #pragma unroll
        for (int r = 0; r < 8; ++r) {
            int row = m0 + kh * 8 + r;
            C[row * DCTX + n0 + j * 16 + n] = c[j][r];
        }
    }
}

// ---------------------------------------------------------------------------
// Fused attention: per (b,s) workgroup.
//   scores[p] = proto[p,:] . r      (q.bk term is softmax-invariant, dropped)
//   w = softmax(scores);  out = sum_p w[p] * proto[p,:]
// The 96 KB proto tile is DMA'd memory->LDS with GLOBAL_LOAD_ASYNC_TO_LDS_B128
// (ASYNCcnt-tracked, no VGPR landing zone), then both passes run out of LDS.
// ---------------------------------------------------------------------------
__global__ __launch_bounds__(256) void fuse_kernel(const float* __restrict__ proto,
                                                   const float* __restrict__ R,
                                                   float* __restrict__ out) {
    extern __shared__ float smem[];
    float* r_s     = smem;                      // 768
    float* proto_s = smem + DCTX;               // 32*768
    float* part_s  = proto_s + Pn * DCTX;       // 256
    float* w_s     = part_s + 256;              // 32

    int s = blockIdx.x;                         // 0 .. BS-1
    int t = threadIdx.x;

    int p   = t >> 3;                           // 0..31
    int sub = t & 7;                            // 0..7, 96 floats each
    const float* src = proto + (size_t)s * Pn * DCTX + p * DCTX + sub * 96;
    float*       dst = proto_s + p * DCTX + sub * 96;

    // issue 24 async 16B mem->LDS copies per thread (ASYNCcnt <= 24 per wave)
    {
        unsigned long long ga = (unsigned long long)(uintptr_t)src;
        unsigned int       la = (unsigned int)(uintptr_t)dst;   // LDS byte offset
        #pragma unroll
        for (int j = 0; j < 24; ++j) {
            asm volatile("global_load_async_to_lds_b128 %0, %1, off"
                         :: "v"(la + j * 16), "v"(ga + (unsigned long long)(j * 16))
                         : "memory");
        }
    }

    // stage r row (768 fp32) into LDS while the async DMA streams
    if (t < 192) ((float4*)r_s)[t] = ((const float4*)(R + (size_t)s * DCTX))[t];

    asm volatile("s_wait_asynccnt 0" ::: "memory");
    __syncthreads();

    // pass 1: score partials out of LDS
    const float* rr = r_s + sub * 96;
    float acc = 0.f;
    #pragma unroll
    for (int j = 0; j < 24; ++j) {
        float4 v  = ((const float4*)dst)[j];
        float4 rv = ((const float4*)rr)[j];
        acc += v.x * rv.x + v.y * rv.y + v.z * rv.z + v.w * rv.w;
    }
    part_s[t] = acc;
    __syncthreads();

    // wave 0: finish scores + softmax over 32 prototypes (wave32 shuffles)
    if (t < 32) {
        float sc = 0.f;
        #pragma unroll
        for (int j = 0; j < 8; ++j) sc += part_s[t * 8 + j];
        float m = sc;
        for (int off = 16; off > 0; off >>= 1) m = fmaxf(m, __shfl_xor(m, off, 32));
        float e = __expf(sc - m);
        float sum = e;
        for (int off = 16; off > 0; off >>= 1) sum += __shfl_xor(sum, off, 32);
        w_s[t] = e / sum;
    }
    __syncthreads();

    // pass 2: weighted sum over prototypes, straight out of LDS (fp32 exact)
    if (t < 192) {
        float4 a4 = {0.f, 0.f, 0.f, 0.f};
        #pragma unroll
        for (int pp = 0; pp < Pn; ++pp) {
            float w = w_s[pp];
            float4 v = ((const float4*)(proto_s + pp * DCTX))[t];
            a4.x += w * v.x; a4.y += w * v.y; a4.z += w * v.z; a4.w += w * v.w;
        }
        ((float4*)(out + (size_t)s * DCTX))[t] = a4;
    }
}

// ---------------------------------------------------------------------------
extern "C" void kernel_launch(void* const* d_in, const int* in_sizes, int n_in,
                              void* d_out, int out_size, void* d_ws, size_t ws_size,
                              hipStream_t stream) {
    const float* sent  = (const float*)d_in[0];   // (B,S,DSENT)
    const float* proto = (const float*)d_in[1];   // (B,S,P,DCTX)
    const float* Wq    = (const float*)d_in[2];   // (DSENT,DCTX)
    const float* bq    = (const float*)d_in[3];   // (DCTX)
    const float* Wk    = (const float*)d_in[4];   // (DCTX,DCTX)
    // bk (d_in[5]) is provably softmax-invariant -> unused.
    float* out = (float*)d_out;                   // (B,S,DCTX) fp32

    char* ws = (char*)d_ws;
    __bf16* sent_bf = (__bf16*)(ws + 0);          //  4,194,304 B
    __bf16* WqT_bf  = (__bf16*)(ws + 4194304);    //  1,572,864 B  [n][k], k contig
    __bf16* Wk_bf   = (__bf16*)(ws + 5767168);    //  1,179,648 B  [c][d], d contig
    __bf16* Q_bf    = (__bf16*)(ws + 6946816);    //  3,145,728 B
    float*  R_ws    = (float*)(ws + 10092544);    //  6,291,456 B  (total 16.0 MB)

    // 1) precision/layout prep
    cvt_bf16_kernel<<<(BS * DSENT) / 256, 256, 0, stream>>>(sent, sent_bf, BS * DSENT);
    transpose_cvt_kernel<<<(DSENT * DCTX) / 256, 256, 0, stream>>>(Wq, WqT_bf,
                                                                   DSENT, DCTX);
    cvt_bf16_kernel<<<(DCTX * DCTX) / 256, 256, 0, stream>>>(Wk, Wk_bf, DCTX * DCTX);

    // 2) Q = sent @ Wq + bq  (bf16 out)  -- 1536 strips / 8 waves per block
    gemm_q_kernel<<<192, 256, 0, stream>>>(sent_bf, WqT_bf, bq, Q_bf);

    // 3) R = Q @ Wk^T  (fp32 out)
    gemm_r_kernel<<<192, 256, 0, stream>>>(Q_bf, Wk_bf, R_ws);

    // 4) fused scores/softmax/weighted-sum, one (b,s) per block
    size_t lds_bytes = (size_t)(DCTX + Pn * DCTX + 256 + 32) * sizeof(float); // ~100 KB
    fuse_kernel<<<BS, 256, lds_bytes, stream>>>(proto, R_ws, out);
}
